// LabeledContrastiveLoss_20701742367088
// MI455X (gfx1250) — matrix operand order repaired and checked by reading, not compiled, not run on hardware
//
#include <hip/hip_runtime.h>
#include <hip/hip_bf16.h>

// ---------------------------------------------------------------------------
// LabeledContrastiveLoss on gfx1250 (MI455X):
//   num = (vi @ vj^T) / 0.5  via v_wmma_f32_16x16x32_f16  (f32 accum)
//   fused streaming base-2 logsumexp epilogue (v_exp_f32 IS exp2 on AMD)
// ---------------------------------------------------------------------------

typedef __attribute__((ext_vector_type(16))) _Float16 v16h;
typedef __attribute__((ext_vector_type(4)))  _Float16 v4h;
typedef __attribute__((ext_vector_type(8)))  float    v8f;

#define B_N    8192
#define D_K    128
#define NCHUNK 8
#define COLS_PER_CHUNK (B_N / NCHUNK)          // 1024
#define TILES_PER_CHUNK (COLS_PER_CHUNK / 16)  // 64

#define TWO_LOG2E 2.8853900817779268f  // (1/TEMP) * log2(e), TEMP = 0.5
#define LN2       0.6931471805599453f
#define NEG_BIG  -30000.0f

__device__ __forceinline__ float ex2(float x) { return __builtin_amdgcn_exp2f(x); }
__device__ __forceinline__ float lg2(float x) { return __builtin_amdgcn_logf(x);  } // base-2

// A-matrix 16x32 f16 fragment: lanes 0-15 row M=m hold K=kb+0..15,
// lanes 16-31 hold K=kb+16..31.  rp points at row start (f32 source).
static __device__ __forceinline__ v16h load_a_frag(const float* __restrict__ rp, int kb, int h) {
  const float* p = rp + kb + 16 * h;
  float4 a0 = *(const float4*)(p);
  float4 a1 = *(const float4*)(p + 4);
  float4 b0 = *(const float4*)(p + 8);
  float4 b1 = *(const float4*)(p + 12);
  v16h f;
  f[0] = (_Float16)a0.x; f[1] = (_Float16)a0.y; f[2]  = (_Float16)a0.z; f[3]  = (_Float16)a0.w;
  f[4] = (_Float16)a1.x; f[5] = (_Float16)a1.y; f[6]  = (_Float16)a1.z; f[7]  = (_Float16)a1.w;
  f[8] = (_Float16)b0.x; f[9] = (_Float16)b0.y; f[10] = (_Float16)b0.z; f[11] = (_Float16)b0.w;
  f[12]= (_Float16)b1.x; f[13]= (_Float16)b1.y; f[14] = (_Float16)b1.z; f[15] = (_Float16)b1.w;
  return f;
}

// B-matrix 32x16 f16 fragment: lane n holds column n; lanes 0-15 K=kb+0..15,
// lanes 16-31 K=kb+16..31 -> one contiguous 32B run per lane from row-major vj (f16).
static __device__ __forceinline__ v16h load_b_frag(const _Float16* __restrict__ vjh,
                                                   int col, int kb, int h) {
  return *(const v16h*)(vjh + (size_t)col * D_K + kb + 16 * h);
}

static __device__ __forceinline__ v8f tile_gemm(const v16h& A0, const v16h& A1,
                                                const v16h& A2, const v16h& A3,
                                                const _Float16* __restrict__ vjh,
                                                int col, int h) {
  v8f c = {0.f, 0.f, 0.f, 0.f, 0.f, 0.f, 0.f, 0.f};
  c = __builtin_amdgcn_wmma_f32_16x16x32_f16(false, A0, false, load_b_frag(vjh, col, 0,  h), (short)0, c, false, false);
  c = __builtin_amdgcn_wmma_f32_16x16x32_f16(false, A1, false, load_b_frag(vjh, col, 32, h), (short)0, c, false, false);
  c = __builtin_amdgcn_wmma_f32_16x16x32_f16(false, A2, false, load_b_frag(vjh, col, 64, h), (short)0, c, false, false);
  c = __builtin_amdgcn_wmma_f32_16x16x32_f16(false, A3, false, load_b_frag(vjh, col, 96, h), (short)0, c, false, false);
  return c;
}

// ---------------------------------------------------------------------------
// K0: convert vj (x[:,0,:]) f32 -> f16 workspace copy; zero the output scalar.
__global__ void lcl_prep(const float* __restrict__ x, _Float16* __restrict__ vjh,
                         float* __restrict__ out) {
  int t = blockIdx.x * blockDim.x + threadIdx.x;
  if (t == 0) out[0] = 0.0f;
  int e = t * 4;                 // element index into [8192 x 128]
  int b = e >> 7, d = e & 127;
  float4 v = *(const float4*)(x + (size_t)b * 256 + d);   // view 0
  v4h o;
  o[0] = (_Float16)v.x; o[1] = (_Float16)v.y; o[2] = (_Float16)v.z; o[3] = (_Float16)v.w;
  *(v4h*)(vjh + e) = o;
}

// ---------------------------------------------------------------------------
// K1: per (16-row block, 1024-col chunk) wave: WMMA tiles + online base-2
// logsumexp over negatives, pos-sum / pos-count; write per-(row,chunk) partials.
__global__ void __launch_bounds__(256) lcl_pass1(
    const float* __restrict__ x, const _Float16* __restrict__ vjh,
    const int* __restrict__ labels,
    float* __restrict__ m_part, float* __restrict__ s_part,
    float* __restrict__ ps_part, float* __restrict__ cn_part) {
  int wid   = (blockIdx.x * 256 + threadIdx.x) >> 5;
  int lane  = threadIdx.x & 31;
  int rb    = wid >> 3;          // row block 0..511
  int chunk = wid & 7;           // column chunk 0..7
  int row_base = rb * 16;
  int col0     = chunk * COLS_PER_CHUNK;
  int h = lane >> 4, m = lane & 15;

  const float* arow = x + (size_t)(row_base + m) * 256 + 128;  // vi = x[:,1,:]
  v16h A0 = load_a_frag(arow, 0,  h);
  v16h A1 = load_a_frag(arow, 32, h);
  v16h A2 = load_a_frag(arow, 64, h);
  v16h A3 = load_a_frag(arow, 96, h);

  int rlab[8];
  float mS[8], sS[8], psS[8], cnS[8];
#pragma unroll
  for (int r = 0; r < 8; ++r) {
    rlab[r] = labels[row_base + 8 * h + r];
    mS[r] = NEG_BIG; sS[r] = 0.f; psS[r] = 0.f; cnS[r] = 0.f;
  }

  for (int t = 0; t < TILES_PER_CHUNK; ++t) {
    int col  = col0 + t * 16 + m;
    int clab = labels[col];
    v8f c = tile_gemm(A0, A1, A2, A3, vjh, col, h);
#pragma unroll
    for (int r = 0; r < 8; ++r) {
      float acc = c[r];
      float y   = acc * TWO_LOG2E;          // num * log2(e), base-2 logits
      bool pos  = (rlab[r] == clab);
      float mo  = mS[r];
      float mn  = pos ? mo : fmaxf(mo, y);  // max only advances on negatives
      float e1  = ex2(mo - mn);             // ==1 unless max advanced
      float e2  = ex2(y - mn);              // discarded by select when pos
      sS[r] = sS[r] * e1 + (pos ? 0.0f : e2);
      mS[r] = mn;
      psS[r] += pos ? acc * 2.0f : 0.0f;    // sum of num over positives
      cnS[r] += pos ? 1.0f : 0.0f;
    }
  }

  // merge the 16 lanes of each half (each holds a disjoint column subset)
#pragma unroll
  for (int r = 0; r < 8; ++r) {
#pragma unroll
    for (int k = 1; k < 16; k <<= 1) {
      float om = __shfl_xor(mS[r], k, 32);
      float os = __shfl_xor(sS[r], k, 32);
      float nm = fmaxf(mS[r], om);
      sS[r] = sS[r] * ex2(mS[r] - nm) + os * ex2(om - nm);
      mS[r] = nm;
      psS[r] += __shfl_xor(psS[r], k, 32);
      cnS[r] += __shfl_xor(cnS[r], k, 32);
    }
  }
  if (m == 0) {
#pragma unroll
    for (int r = 0; r < 8; ++r) {
      int row = row_base + 8 * h + r;
      m_part [row * NCHUNK + chunk] = mS[r];
      s_part [row * NCHUNK + chunk] = sS[r];
      ps_part[row * NCHUNK + chunk] = psS[r];
      cn_part[row * NCHUNK + chunk] = cnS[r];
    }
  }
}

// ---------------------------------------------------------------------------
// K2: merge 8 chunk partials per row -> yL = log2(sum 2^y over negatives);
// also totals for pos-sum / pos-count, and zero the denom accumulators.
__global__ void lcl_combine(const float* __restrict__ m_part, const float* __restrict__ s_part,
                            const float* __restrict__ ps_part, const float* __restrict__ cn_part,
                            float* __restrict__ yL, float* __restrict__ ps_row,
                            float* __restrict__ cn_row, float* __restrict__ denom) {
  int row = blockIdx.x * blockDim.x + threadIdx.x;
  float mM = -3.0e38f;
#pragma unroll
  for (int k = 0; k < NCHUNK; ++k) mM = fmaxf(mM, m_part[row * NCHUNK + k]);
  float s = 0.f, ps = 0.f, cn = 0.f;
#pragma unroll
  for (int k = 0; k < NCHUNK; ++k) {
    s  += s_part [row * NCHUNK + k] * ex2(m_part[row * NCHUNK + k] - mM);
    ps += ps_part[row * NCHUNK + k];
    cn += cn_part[row * NCHUNK + k];
  }
  yL[row]     = mM + lg2(s);   // -inf if no negatives (handled downstream)
  ps_row[row] = ps;
  cn_row[row] = cn;
  denom[row]  = 0.0f;
}

// ---------------------------------------------------------------------------
// K3: recompute tiles; accumulate sum over positives of logaddexp(num, L_neg)
//   = ln2 * (max(y,yL) + log2(1 + 2^-|y-yL|))
__global__ void __launch_bounds__(256) lcl_pass2(
    const float* __restrict__ x, const _Float16* __restrict__ vjh,
    const int* __restrict__ labels, const float* __restrict__ yL,
    float* __restrict__ denom) {
  int wid   = (blockIdx.x * 256 + threadIdx.x) >> 5;
  int lane  = threadIdx.x & 31;
  int rb    = wid >> 3;
  int chunk = wid & 7;
  int row_base = rb * 16;
  int col0     = chunk * COLS_PER_CHUNK;
  int h = lane >> 4, m = lane & 15;

  const float* arow = x + (size_t)(row_base + m) * 256 + 128;
  v16h A0 = load_a_frag(arow, 0,  h);
  v16h A1 = load_a_frag(arow, 32, h);
  v16h A2 = load_a_frag(arow, 64, h);
  v16h A3 = load_a_frag(arow, 96, h);

  int rlab[8]; float yLr[8], dA[8];
#pragma unroll
  for (int r = 0; r < 8; ++r) {
    int row = row_base + 8 * h + r;
    rlab[r] = labels[row];
    yLr[r]  = yL[row];
    dA[r]   = 0.f;
  }

  for (int t = 0; t < TILES_PER_CHUNK; ++t) {
    int col  = col0 + t * 16 + m;
    int clab = labels[col];
    v8f c = tile_gemm(A0, A1, A2, A3, vjh, col, h);
#pragma unroll
    for (int r = 0; r < 8; ++r) {
      float y   = c[r] * TWO_LOG2E;
      bool pos  = (rlab[r] == clab);
      float Mv  = fmaxf(y, yLr[r]);
      float tt  = ex2(-__builtin_fabsf(y - yLr[r]));   // 0 when yL = -inf
      float lae = LN2 * (Mv + lg2(1.0f + tt));
      dA[r] += pos ? lae : 0.0f;
    }
  }

#pragma unroll
  for (int r = 0; r < 8; ++r) {
#pragma unroll
    for (int k = 1; k < 16; k <<= 1) dA[r] += __shfl_xor(dA[r], k, 32);
  }
  if (m == 0) {
#pragma unroll
    for (int r = 0; r < 8; ++r)
      atomicAdd(&denom[row_base + 8 * h + r], dA[r]);
  }
}

// ---------------------------------------------------------------------------
// K4: a = (denom - pos_sum)/cnt, mean over rows into out[0].
__global__ void lcl_finalize(const float* __restrict__ denom, const float* __restrict__ ps_row,
                             const float* __restrict__ cn_row, float* __restrict__ out) {
  __shared__ float red[256];
  int row = blockIdx.x * 256 + threadIdx.x;
  float a = (denom[row] - ps_row[row]) / cn_row[row];
  red[threadIdx.x] = a * (1.0f / (float)B_N);
  __syncthreads();
  for (int s = 128; s > 0; s >>= 1) {
    if (threadIdx.x < s) red[threadIdx.x] += red[threadIdx.x + s];
    __syncthreads();
  }
  if (threadIdx.x == 0) atomicAdd(out, red[0]);
}

// ---------------------------------------------------------------------------
extern "C" void kernel_launch(void* const* d_in, const int* in_sizes, int n_in,
                              void* d_out, int out_size, void* d_ws, size_t ws_size,
                              hipStream_t stream) {
  const float* x      = (const float*)d_in[0];
  const int*   labels = (const int*)d_in[1];
  float*       out    = (float*)d_out;

  // workspace layout (~3.2 MB total)
  char* ws = (char*)d_ws;
  _Float16* vjh    = (_Float16*)ws;                                  // 8192*128 f16 = 2 MB
  float*    m_part = (float*)(ws + (size_t)B_N * D_K * sizeof(_Float16));
  float*    s_part = m_part  + B_N * NCHUNK;
  float*    ps_part= s_part  + B_N * NCHUNK;
  float*    cn_part= ps_part + B_N * NCHUNK;
  float*    yL     = cn_part + B_N * NCHUNK;
  float*    ps_row = yL     + B_N;
  float*    cn_row = ps_row + B_N;
  float*    denom  = cn_row + B_N;

  const int waves  = (B_N / 16) * NCHUNK;       // 4096 waves
  const int blocks = waves * 32 / 256;          // 512 blocks of 8 waves

  lcl_prep    <<<(B_N * D_K / 4) / 256, 256, 0, stream>>>(x, vjh, out);
  lcl_pass1   <<<blocks, 256, 0, stream>>>(x, vjh, labels, m_part, s_part, ps_part, cn_part);
  lcl_combine <<<B_N / 256, 256, 0, stream>>>(m_part, s_part, ps_part, cn_part, yL, ps_row, cn_row, denom);
  lcl_pass2   <<<blocks, 256, 0, stream>>>(x, vjh, labels, yL, denom);
  lcl_finalize<<<B_N / 256, 256, 0, stream>>>(denom, ps_row, cn_row, out);
}
